// CudaGRULM_4887672783104
// MI455X (gfx1250) — matrix operand order repaired and checked by reading, not compiled, not run on hardware
//
#include <hip/hip_runtime.h>

// ---------------- CDNA5 WMMA types ----------------
typedef __attribute__((ext_vector_type(16))) __bf16 v16bf;
typedef __attribute__((ext_vector_type(8)))  __bf16 v8bf;
typedef __attribute__((ext_vector_type(8)))  float  v8f;

constexpr int V_     = 256;
constexpr int D_     = 512;
constexpr int DEPTH_ = 12;
constexpr int DI_    = 512;
constexpr int B_     = 16;
constexpr int T_     = 2048;
constexpr int BT_    = B_ * T_;   // 32768

static __device__ __forceinline__ int bt2tb(int row) {
  // row = b*T + t  ->  t*B + b   (T = 2048 = 2^11)
  int b = row >> 11;
  int t = row & (T_ - 1);
  return t * B_ + b;
}

static __device__ __forceinline__ v8f wmma_bf16(v16bf a, v16bf b, v8f c) {
  // D = A(16x32 bf16) * B(32x16 bf16) + C(16x16 f32)
  return __builtin_amdgcn_wmma_f32_16x16x32_bf16(
      /*neg_a=*/false, a, /*neg_b=*/false, b,
      /*c_mod=*/(short)0, c, /*reuse_a=*/false, /*reuse_b=*/false);
}

// ---- CDNA5 async global->LDS copy (ASYNCcnt path, ISA §10.7 / §15.18) ----
// VDST = per-lane LDS byte address (low 32 bits of flat shared pointer:
// the LDS aperture keys on addr[63:32], addr[31:0] is the LDS offset).
static __device__ __forceinline__ void async_cp_b128(unsigned lds_off,
                                                     const void* gptr) {
  asm volatile("global_load_async_to_lds_b128 %0, %1, off"
               :: "v"(lds_off), "v"((unsigned long long)(size_t)gptr)
               : "memory");
}
static __device__ __forceinline__ void wait_asynccnt0() {
  asm volatile("s_wait_asynccnt 0" ::: "memory");
}

// A-matrix 16x32 bf16 tile load (row-major source, per-lane row pointer).
// Lanes 0-15: row=lane,   K [k0..k0+15]
// Lanes 16-31: row=lane-16, K [k0+16..k0+31]
static __device__ __forceinline__ v16bf load_a16(const __bf16* rowptr, int k0, int lane) {
  int ko = k0 + ((lane >> 4) << 4);
  v8bf lo = *(const v8bf*)(rowptr + ko);
  v8bf hi = *(const v8bf*)(rowptr + ko + 8);
  return __builtin_shufflevector(lo, hi, 0, 1, 2, 3, 4, 5, 6, 7,
                                         8, 9, 10, 11, 12, 13, 14, 15);
}

// B-matrix 32x16 bf16 tile load: a lane's 16 K-values for its column n are
// contiguous (source row-major (N,K), B = W^T).
static __device__ __forceinline__ v16bf load_b16(const __bf16* Wrow /* = base + n*ld */,
                                                 int k0, int lane) {
  int kk = k0 + ((lane >> 4) << 4);
  return *(const v16bf*)(Wrow + kk);
}

// ---------------- elementwise kernels ----------------
__global__ void f2bf_kernel(const float* __restrict__ in, __bf16* __restrict__ out, int n) {
  int i = blockIdx.x * blockDim.x + threadIdx.x;
  if (i < n) out[i] = (__bf16)in[i];
}

__global__ void embed_kernel(const int* __restrict__ x, const float* __restrict__ emb,
                             float* __restrict__ h) {
  const int bt  = blockIdx.x;          // b*T + t
  const int tok = x[bt];
  const float* e = emb + (size_t)tok * D_;
  float* o = h + (size_t)bt * D_;
  for (int i = threadIdx.x; i < D_; i += blockDim.x) o[i] = e[i];
}

__global__ __launch_bounds__(256)
void layernorm_kernel(const float* __restrict__ x, const float* __restrict__ gamma,
                      const float* __restrict__ beta, __bf16* __restrict__ out) {
  const int row  = blockIdx.x;
  const float* xr = x + (size_t)row * D_;
  const int tid = threadIdx.x, lane = tid & 31, wv = tid >> 5;
  float s = 0.f, s2 = 0.f;
  for (int i = tid; i < D_; i += 256) { float v = xr[i]; s += v; s2 += v * v; }
#pragma unroll
  for (int off = 16; off > 0; off >>= 1) {
    s  += __shfl_down(s, off);
    s2 += __shfl_down(s2, off);
  }
  __shared__ float red[2][8];
  if (lane == 0) { red[0][wv] = s; red[1][wv] = s2; }
  __syncthreads();
  float S = 0.f, S2 = 0.f;
#pragma unroll
  for (int w = 0; w < 8; ++w) { S += red[0][w]; S2 += red[1][w]; }
  const float mean = S * (1.f / D_);
  const float var  = S2 * (1.f / D_) - mean * mean;
  const float inv  = rsqrtf(var + 1e-5f);
  __bf16* orow = out + (size_t)row * D_;
  for (int i = tid; i < D_; i += 256)
    orow[i] = (__bf16)((xr[i] - mean) * inv * gamma[i] + beta[i]);
}

// ---------------- LDS-blocked WMMA GEMM with async double buffering -------
// C[crow(m), n] (+=) sum_k A[arow(m), k] * W[n, k]  (+ bias[n])
// remap: 0 none | 1 crow = bt2tb(row) | 2 arow = bt2tb(row)
// Block tile 128x128; 8 waves in 2(M) x 4(N); wave tile 64x32.
// K-slices DMA'd to LDS with global_load_async_to_lds_b128 (double buffer),
// one barrier per K-step; 64 FLOP per global byte.
__global__ __launch_bounds__(256)
void gemm_wmma_kernel(const __bf16* __restrict__ A, int lda,
                      const __bf16* __restrict__ W,
                      const float* __restrict__ bias,
                      float* __restrict__ outF, __bf16* __restrict__ outB,
                      int ldc, int M, int N, int K, int remap, int accum) {
  constexpr int MB = 128, NB = 128, KB = 32, LDP = 48;  // 96B padded row stride
  __shared__ __bf16 Ap[2][MB * LDP];
  __shared__ __bf16 Bp[2][NB * LDP];

  const int tid  = threadIdx.x;
  const int lane = tid & 31;
  const int wv   = tid >> 5;         // 0..7
  const int wm   = wv & 1;           // wave M position (0..1)
  const int wn   = wv >> 1;          // wave N position (0..3)
  const int bm   = blockIdx.x * MB;
  const int bn   = blockIdx.y * NB;

  // cooperative staging: each thread owns one 32B chunk of A and of B
  const int ldr = tid >> 1;          // panel row 0..127
  const int ldo = (tid & 1) << 4;    // half-offset 0 or 16
  int arow = bm + ldr;
  if (remap == 2) arow = bt2tb(arow);
  const __bf16* agp = A + (size_t)arow * lda + ldo;
  const __bf16* bgp = W + (size_t)(bn + ldr) * K + ldo;
  const unsigned aoff0 = (unsigned)(size_t)(&Ap[0][ldr * LDP + ldo]);
  const unsigned aoff1 = (unsigned)(size_t)(&Ap[1][ldr * LDP + ldo]);
  const unsigned boff0 = (unsigned)(size_t)(&Bp[0][ldr * LDP + ldo]);
  const unsigned boff1 = (unsigned)(size_t)(&Bp[1][ldr * LDP + ldo]);

  const int ml = lane & 15;
  const int apwo = (wm * 64 + ml) * LDP;   // + i*16*LDP per sub-tile
  const int bpwo = (wn * 32 + ml) * LDP;   // + j*16*LDP per sub-tile

  v8f acc[4][2] = {};

  auto stage = [&](int buf, int k) {
    unsigned la = buf ? aoff1 : aoff0;
    unsigned lb = buf ? boff1 : boff0;
    async_cp_b128(la,      agp + k);        // 16B
    async_cp_b128(la + 16, agp + k + 8);    // next 16B
    async_cp_b128(lb,      bgp + k);
    async_cp_b128(lb + 16, bgp + k + 8);
  };

  // prologue: DMA K-slice 0 into buffer 0
  stage(0, 0);

  int cur = 0;
  for (int k0 = 0; k0 < K; k0 += KB, cur ^= 1) {
    wait_asynccnt0();                // this wave's DMA into buf[cur] landed
    __syncthreads();                 // everyone's landed; buf[cur^1] free
    if (k0 + KB < K) stage(cur ^ 1, k0 + KB);  // overlap DMA with WMMAs

    const __bf16* apw = &Ap[cur][apwo];
    const __bf16* bpw = &Bp[cur][bpwo];
#pragma unroll
    for (int i = 0; i < 4; ++i) {
      v16bf a = load_a16(apw + i * (16 * LDP), 0, lane);
#pragma unroll
      for (int j = 0; j < 2; ++j) {
        v16bf b = load_b16(bpw + j * (16 * LDP), 0, lane);
        acc[i][j] = wmma_bf16(a, b, acc[i][j]);
      }
    }
  }

  // epilogue: C layout lane=(N, halves), VGPR r = M row
#pragma unroll
  for (int j = 0; j < 2; ++j) {
    const int n  = bn + wn * 32 + j * 16 + ml;
    const float bvs = bias ? bias[n] : 0.f;
#pragma unroll
    for (int i = 0; i < 4; ++i) {
      const int mb0 = bm + wm * 64 + i * 16 + ((lane >> 4) << 3);
#pragma unroll
      for (int r = 0; r < 8; ++r) {
        int row  = mb0 + r;
        int crow = (remap == 1) ? bt2tb(row) : row;
        size_t idx = (size_t)crow * ldc + n;
        float v = acc[i][j][r] + bvs;
        if (accum && outF) v += outF[idx];
        if (outF) outF[idx] = v;
        if (outB) outB[idx] = (__bf16)v;
      }
    }
  }
}

// ---------------- persistent GRU scan ----------------
// One workgroup, 32 waves. h (16x512) lives in LDS as bf16.
// Wave w owns output columns [16w, 16w+16): computes z tile, r tile,
// writes r*h to LDS, then h_tilde tile, then the gated update.
__global__ __launch_bounds__(1024)
void gru_scan_kernel(const __bf16* __restrict__ xzr,  // [T*B, 2*DI] (bias folded in)
                     const __bf16* __restrict__ xh,   // [T*B, DI]   (bias folded in)
                     const __bf16* __restrict__ Uzr,  // [2*DI, DI] bf16
                     const __bf16* __restrict__ Uh,   // [DI, DI]  bf16
                     __bf16* __restrict__ hs) {       // [T*B, DI]
  __shared__ __bf16 h_lds[B_ * DI_];
  __shared__ __bf16 rh_lds[B_ * DI_];

  const int tid  = threadIdx.x;
  const int lane = tid & 31;
  const int wv   = tid >> 5;                 // 0..31
  const int nl   = (wv << 4) + (lane & 15);  // this lane's column in [0,512)
  const int mb   = (lane >> 4) << 3;         // C rows base: 0 or 8
  const int kofsB = (lane >> 4) << 4;

  const __bf16* Uz_row = Uzr + (size_t)nl * DI_;
  const __bf16* Ur_row = Uzr + (size_t)(DI_ + nl) * DI_;
  const __bf16* Uh_row = Uh  + (size_t)nl * DI_;
  const __bf16* hrow   = h_lds  + (size_t)(lane & 15) * DI_;
  const __bf16* rhrow  = rh_lds + (size_t)(lane & 15) * DI_;

  for (int i = tid; i < B_ * DI_; i += 1024) h_lds[i] = (__bf16)0.f;
  __syncthreads();

#pragma unroll 1
  for (int t = 0; t < T_; ++t) {
    // prefetch next timestep's activations (global_prefetch_b8) to hide the
    // serial-critical-path VMEM latency behind this step's WMMAs
    if (t + 1 < T_) {
      const __bf16* nzr = xzr + (size_t)(t + 1) * (B_ * 2 * DI_);
      const __bf16* nxh = xh  + (size_t)(t + 1) * (B_ * DI_);
      __builtin_prefetch(nzr + mb * (2 * DI_) + nl, 0, 3);
      __builtin_prefetch(nzr + mb * (2 * DI_) + DI_ + nl, 0, 3);
      __builtin_prefetch(nxh + mb * DI_ + nl, 0, 3);
    }

    // ---- zr = sigmoid(xzr_t + h @ Uzr^T) ----
    v8f cz = {};
    v8f cr = {};
#pragma unroll
    for (int k0 = 0; k0 < DI_; k0 += 32) {
      v16bf a  = load_a16(hrow, k0, lane);
      v16bf bz = *(const v16bf*)(Uz_row + k0 + kofsB);
      v16bf br = *(const v16bf*)(Ur_row + k0 + kofsB);
      cz = wmma_bf16(a, bz, cz);
      cr = wmma_bf16(a, br, cr);
    }
    const __bf16* xzr_t = xzr + (size_t)t * (B_ * 2 * DI_);
    float zv[8];
#pragma unroll
    for (int r = 0; r < 8; ++r) {
      int m = mb + r;
      float pz = cz[r] + (float)xzr_t[m * (2 * DI_) + nl];
      float pr = cr[r] + (float)xzr_t[m * (2 * DI_) + DI_ + nl];
      float z  = 1.f / (1.f + __expf(-pz));
      float rg = 1.f / (1.f + __expf(-pr));
      zv[r] = z;
      rh_lds[m * DI_ + nl] = (__bf16)(rg * (float)h_lds[m * DI_ + nl]);
    }
    __syncthreads();  // rh visible to all waves; old-h readers are done

    // ---- h_tilde = tanh(xh_t + (r*h) @ Uh^T) ----
    v8f ch = {};
#pragma unroll
    for (int k0 = 0; k0 < DI_; k0 += 32) {
      v16bf a  = load_a16(rhrow, k0, lane);
      v16bf bh = *(const v16bf*)(Uh_row + k0 + kofsB);
      ch = wmma_bf16(a, bh, ch);
    }
    const __bf16* xh_t = xh + (size_t)t * (B_ * DI_);
    __bf16* hs_t = hs + (size_t)t * (B_ * DI_);
#pragma unroll
    for (int r = 0; r < 8; ++r) {
      int m  = mb + r;
      float ht = tanhf(ch[r] + (float)xh_t[m * DI_ + nl]);
      float hv = (float)h_lds[m * DI_ + nl];          // own column only
      float hn = (1.f - zv[r]) * hv + zv[r] * ht;
      __bf16 hb = (__bf16)hn;
      h_lds[m * DI_ + nl] = hb;                        // own column only
      hs_t[m * DI_ + nl]  = hb;
    }
    __syncthreads();  // new h visible before next step's zr matmul
  }
}

// ---------------- host driver ----------------
extern "C" void kernel_launch(void* const* d_in, const int* in_sizes, int n_in,
                              void* d_out, int out_size, void* d_ws, size_t ws_size,
                              hipStream_t stream) {
  (void)in_sizes; (void)n_in; (void)out_size; (void)ws_size;

  const int*   x    = (const int*)  d_in[0];
  const float* emb  = (const float*)d_in[1];
  const float* ln_g = (const float*)d_in[2];
  const float* ln_b = (const float*)d_in[3];
  const float* Win  = (const float*)d_in[4];
  const float* W_zr = (const float*)d_in[5];
  const float* U_zr = (const float*)d_in[6];
  const float* W_h  = (const float*)d_in[7];
  const float* U_h  = (const float*)d_in[8];
  const float* b_zr = (const float*)d_in[9];
  const float* b_h  = (const float*)d_in[10];
  const float* Wout = (const float*)d_in[11];
  const float* n_g  = (const float*)d_in[12];
  const float* n_b  = (const float*)d_in[13];

  char* wsp = (char*)d_ws;
  size_t off = 0;
  auto alloc = [&](size_t bytes) -> void* {
    void* p = wsp + off;
    off += (bytes + 255) & ~(size_t)255;
    return p;
  };

  float*  h       = (float*) alloc((size_t)BT_ * D_ * sizeof(float));
  __bf16* hn_bf   = (__bf16*)alloc((size_t)BT_ * D_ * sizeof(__bf16));
  __bf16* hp_bf   = (__bf16*)alloc((size_t)BT_ * DI_ * sizeof(__bf16));
  __bf16* xzr_bf  = (__bf16*)alloc((size_t)BT_ * 2 * DI_ * sizeof(__bf16));
  __bf16* xh_bf   = (__bf16*)alloc((size_t)BT_ * DI_ * sizeof(__bf16));
  __bf16* hs_bf   = (__bf16*)alloc((size_t)BT_ * DI_ * sizeof(__bf16));
  __bf16* emb_bf  = (__bf16*)alloc((size_t)V_ * D_ * sizeof(__bf16));
  __bf16* win_bf  = (__bf16*)alloc((size_t)DI_ * D_ * sizeof(__bf16));
  __bf16* wzr_bf  = (__bf16*)alloc((size_t)2 * DI_ * DI_ * sizeof(__bf16));
  __bf16* uzr_bf  = (__bf16*)alloc((size_t)2 * DI_ * DI_ * sizeof(__bf16));
  __bf16* wh_bf   = (__bf16*)alloc((size_t)DI_ * DI_ * sizeof(__bf16));
  __bf16* uh_bf   = (__bf16*)alloc((size_t)DI_ * DI_ * sizeof(__bf16));
  __bf16* wout_bf = (__bf16*)alloc((size_t)D_ * DI_ * sizeof(__bf16));

  auto cvt = [&](const float* src, __bf16* dst, int n) {
    f2bf_kernel<<<(n + 255) / 256, 256, 0, stream>>>(src, dst, n);
  };
  auto gemm = [&](const __bf16* A, int lda, const __bf16* W, const float* bias,
                  float* oF, __bf16* oB, int ldc, int M, int N, int K,
                  int remap, int accum) {
    dim3 grid(M / 128, N / 128);
    gemm_wmma_kernel<<<grid, 256, 0, stream>>>(A, lda, W, bias, oF, oB, ldc,
                                               M, N, K, remap, accum);
  };

  // h = embedding[x]
  embed_kernel<<<BT_, 256, 0, stream>>>(x, emb, h);
  cvt(emb, emb_bf, V_ * D_);

  for (int i = 0; i < DEPTH_; ++i) {
    cvt(Win  + (size_t)i * DI_ * D_,      win_bf,  DI_ * D_);
    cvt(W_zr + (size_t)i * 2 * DI_ * DI_, wzr_bf,  2 * DI_ * DI_);
    cvt(U_zr + (size_t)i * 2 * DI_ * DI_, uzr_bf,  2 * DI_ * DI_);
    cvt(W_h  + (size_t)i * DI_ * DI_,     wh_bf,   DI_ * DI_);
    cvt(U_h  + (size_t)i * DI_ * DI_,     uh_bf,   DI_ * DI_);
    cvt(Wout + (size_t)i * D_ * DI_,      wout_bf, D_ * DI_);

    // hn = LN(h)
    layernorm_kernel<<<BT_, 256, 0, stream>>>(h, ln_g + (size_t)i * D_,
                                              ln_b + (size_t)i * D_, hn_bf);
    // hp[t*B+b, e] = hn[b*T+t, :] @ Win^T   (remap output rows bt -> tb)
    gemm(hn_bf, D_, win_bf, nullptr, nullptr, hp_bf, DI_, BT_, DI_, D_, 1, 0);
    // xzr = hp @ W_zr^T + b_zr ; xh = hp @ W_h^T + b_h
    gemm(hp_bf, DI_, wzr_bf, b_zr + (size_t)i * 2 * DI_, nullptr, xzr_bf,
         2 * DI_, BT_, 2 * DI_, DI_, 0, 0);
    gemm(hp_bf, DI_, wh_bf,  b_h  + (size_t)i * DI_,     nullptr, xh_bf,
         DI_, BT_, DI_, DI_, 0, 0);
    // sequential GRU scan (persistent single workgroup, WMMA recurrence)
    gru_scan_kernel<<<1, 1024, 0, stream>>>(xzr_bf, xh_bf, uzr_bf, uh_bf, hs_bf);
    // h[b*T+t, d] += hs[t*B+b, :] @ Wout^T  (remap A rows, accumulate residual)
    gemm(hs_bf, DI_, wout_bf, nullptr, h, nullptr, D_, BT_, D_, DI_, 2, 1);
  }

  // final LN + logits = hn @ embedding^T
  layernorm_kernel<<<BT_, 256, 0, stream>>>(h, n_g, n_b, hn_bf);
  gemm(hn_bf, D_, emb_bf, nullptr, (float*)d_out, nullptr, V_, BT_, V_, D_, 0, 0);
}
